// BiLSTMClassifier_11063835755286
// MI455X (gfx1250) — matrix operand order repaired and checked by reading, not compiled, not run on hardware
//
#include <hip/hip_runtime.h>
#include <hip/hip_bf16.h>

#define VOCAB 32000
#define EMB   128
#define HID   256
#define NCLS  8
#define BB    256
#define SS    512
#define KTOT  384   // EMB + HID (concatenated GEMM K)
#define NTOT  1024  // 4*HID gate columns
#define XH_STRIDE 392  // bf16 elems per LDS row: 384 + 8 pad (row = 784B, 16B aligned)

typedef __attribute__((ext_vector_type(16))) __bf16 v16bf;
typedef __attribute__((ext_vector_type(8)))  float  v8f;
typedef __attribute__((ext_vector_type(4)))  unsigned int v4u;

union FragAB { v16bf v; v4u q[2]; };

__device__ __forceinline__ unsigned short f2bf(float f) {
  unsigned int u = __float_as_uint(f);
  u += 0x7FFFu + ((u >> 16) & 1u);   // round-to-nearest-even
  return (unsigned short)(u >> 16);
}

#if __has_builtin(__builtin_amdgcn_rcpf)
__device__ __forceinline__ float frcp(float x) { return __builtin_amdgcn_rcpf(x); }
#else
__device__ __forceinline__ float frcp(float x) { return 1.0f / x; }
#endif

// sigmoid via v_exp_f32 + v_rcp_f32 (no IEEE division sequences on the scan path)
__device__ __forceinline__ float sigm(float x) { return frcp(1.0f + __expf(-x)); }

#if __has_builtin(__builtin_amdgcn_tanhf)
__device__ __forceinline__ float ftanh(float x) { return __builtin_amdgcn_tanhf(x); }
#else
__device__ __forceinline__ float ftanh(float x) {
  float e = __expf(2.0f * x);
  return (e - 1.0f) * frcp(e + 1.0f);
}
#endif

// ---- fp32 -> bf16 elementwise (embedding table) ----
__global__ void cvt_f32_bf16(const float* __restrict__ src,
                             unsigned short* __restrict__ dst, int n) {
  int i = blockIdx.x * blockDim.x + threadIdx.x;
  if (i < n) dst[i] = f2bf(src[i]);
}

// ---- pack [Wi;Wh] ([K,N] fp32) into [N][K] bf16 (K-contiguous per column) ----
__global__ void pack_w(const float* __restrict__ Wi, const float* __restrict__ Wh,
                       unsigned short* __restrict__ dst) {
  int i = blockIdx.x * blockDim.x + threadIdx.x;  // over NTOT*KTOT
  if (i >= NTOT * KTOT) return;
  int n = i / KTOT, k = i % KTOT;
  float v = (k < EMB) ? Wi[k * NTOT + n] : Wh[(k - EMB) * NTOT + n];
  dst[i] = f2bf(v);
}

// ---- fused embed + input-proj + LSTM recurrence, one batch-tile x direction per WG ----
__global__ void __launch_bounds__(512, 1)
bilstm_kernel(const int* __restrict__ tokens,
              const unsigned short* __restrict__ embb,  // [VOCAB][EMB] bf16
              const unsigned short* __restrict__ wcat,  // [2][NTOT][KTOT] bf16
              const float* __restrict__ b_f,
              const float* __restrict__ b_b,
              float* __restrict__ feat)                 // [BB][2*HID] fp32
{
  __shared__ __align__(16) unsigned short xh[16 * XH_STRIDE];  // [m][ x(0:128) | h(128:384) ]

  const int wg   = blockIdx.x;        // 32 WGs: 16 batch tiles x 2 directions
  const int dir  = wg & 1;
  const int m0   = (wg >> 1) * 16;
  const int tid  = threadIdx.x;
  const int wave = tid >> 5;          // 0..15
  const int lane = tid & 31;
  const int hlf  = lane >> 4;         // lane half (K/M split per ISA layout)
  const int ln   = lane & 15;

  const float* bias = dir ? b_b : b_f;
  const unsigned short* W = wcat + (size_t)dir * NTOT * KTOT;

  // Per-wave gate columns: n(g) = g*HID + wave*16 + ln  -> i/f/g/o of same hidden
  // unit live in this wave's accumulators (no LDS z exchange needed).
  float bias_v[4];
  const unsigned short* bp[4];
#pragma unroll
  for (int g = 0; g < 4; ++g) {
    int n = g * HID + wave * 16 + ln;
    bias_v[g] = bias[n];
    bp[g] = W + (size_t)n * KTOT + hlf * 16;  // per-lane column base, K half selected
  }
  const unsigned short* arow = &xh[ln * XH_STRIDE + hlf * 16];  // per-lane A row base

  // zero h region (cols 128..383, all 16 rows): 4096 bf16, 8 per thread
  {
    int idx = tid * 8;
    int m = idx >> 8, c = idx & 255;
    v4u z = {0u, 0u, 0u, 0u};
    *(v4u*)&xh[m * XH_STRIDE + 128 + c] = z;
  }
  // load x(0) embedding rows (16 rows x 128 bf16)
  if (tid < 256) {
    int m = tid >> 4, seg = tid & 15;
    int tt = dir ? (SS - 1) : 0;
    int tok = tokens[(m0 + m) * SS + tt];
    *(v4u*)&xh[m * XH_STRIDE + seg * 8] =
        *(const v4u*)&embb[(size_t)tok * EMB + seg * 8];
  }
  __syncthreads();

  v8f cfrag = {};   // persistent cell state: (M = r + hlf*8, N = wave*16 + ln)

  for (int t = 0; t < SS; ++t) {
    v8f acc0, acc1, acc2, acc3;
#pragma unroll
    for (int e = 0; e < 8; ++e) {
      acc0[e] = bias_v[0]; acc1[e] = bias_v[1];
      acc2[e] = bias_v[2]; acc3[e] = bias_v[3];
    }

    // z = [x|h] @ [Wi;Wh] : K = 384 = 12 k-tiles of 32, bf16 WMMA, f32 accum.
    // unroll 2: enough independent registers to overlap loads with WMMA, no spills.
#pragma unroll 2
    for (int kt = 0; kt < 12; ++kt) {
      const int ko = kt * 32;
      FragAB a, b0, b1, b2, b3;
      // issue all loads first so they form clauses and overlap prior WMMAs
      a.q[0]  = *(const v4u*)(arow + ko);
      a.q[1]  = *(const v4u*)(arow + ko + 8);
      b0.q[0] = *(const v4u*)(bp[0] + ko);
      b0.q[1] = *(const v4u*)(bp[0] + ko + 8);
      b1.q[0] = *(const v4u*)(bp[1] + ko);
      b1.q[1] = *(const v4u*)(bp[1] + ko + 8);
      b2.q[0] = *(const v4u*)(bp[2] + ko);
      b2.q[1] = *(const v4u*)(bp[2] + ko + 8);
      b3.q[0] = *(const v4u*)(bp[3] + ko);
      b3.q[1] = *(const v4u*)(bp[3] + ko + 8);
      acc0 = __builtin_amdgcn_wmma_f32_16x16x32_bf16(false, a.v, false, b0.v,
                                                     (short)0, acc0, false, false);
      acc1 = __builtin_amdgcn_wmma_f32_16x16x32_bf16(false, a.v, false, b1.v,
                                                     (short)0, acc1, false, false);
      acc2 = __builtin_amdgcn_wmma_f32_16x16x32_bf16(false, a.v, false, b2.v,
                                                     (short)0, acc2, false, false);
      acc3 = __builtin_amdgcn_wmma_f32_16x16x32_bf16(false, a.v, false, b3.v,
                                                     (short)0, acc3, false, false);
    }
    __syncthreads();   // all LDS reads of x(t), h(t) complete

    // Gate math fully in registers (C-layout: elem r -> M = r + hlf*8, N = ln)
    v8f hnew;
#pragma unroll
    for (int r = 0; r < 8; ++r) {
      float c = sigm(acc1[r]) * cfrag[r] + sigm(acc0[r]) * ftanh(acc2[r]);
      cfrag[r] = c;
      hnew[r] = sigm(acc3[r]) * ftanh(c);
    }
    // write h(t+1) back to LDS (this wave exclusively owns cols 128+wave*16..+15)
    {
      unsigned short* hp = &xh[(hlf * 8) * XH_STRIDE + 128 + wave * 16 + ln];
#pragma unroll
      for (int r = 0; r < 8; ++r) hp[r * XH_STRIDE] = f2bf(hnew[r]);
    }
    // stage x(t+1)
    if (t + 1 < SS && tid < 256) {
      int m = tid >> 4, seg = tid & 15;
      int tt = dir ? (SS - 2 - t) : (t + 1);
      int tok = tokens[(m0 + m) * SS + tt];
      *(v4u*)&xh[m * XH_STRIDE + seg * 8] =
          *(const v4u*)&embb[(size_t)tok * EMB + seg * 8];
    }
    __syncthreads();   // writes visible before next step's GEMM
  }

  // final cell state -> feature buffer [BB][2H]
  float* fp = feat + (size_t)(m0 + hlf * 8) * (2 * HID) + dir * HID + wave * 16 + ln;
#pragma unroll
  for (int r = 0; r < 8; ++r) fp[(size_t)r * 2 * HID] = cfrag[r];
}

// ---- tiny classifier head: out[256,8] = feat[256,512] @ Wd + bd ----
__global__ void head_kernel(const float* __restrict__ feat,
                            const float* __restrict__ Wd,
                            const float* __restrict__ bd,
                            float* __restrict__ out) {
  int b = blockIdx.x * blockDim.x + threadIdx.x;
  if (b >= BB) return;
  float acc[NCLS];
#pragma unroll
  for (int n = 0; n < NCLS; ++n) acc[n] = bd[n];
  const float* fb = feat + (size_t)b * 2 * HID;
  for (int k = 0; k < 2 * HID; ++k) {
    float f = fb[k];
#pragma unroll
    for (int n = 0; n < NCLS; ++n) acc[n] += f * Wd[k * NCLS + n];
  }
#pragma unroll
  for (int n = 0; n < NCLS; ++n) out[b * NCLS + n] = acc[n];
}

extern "C" void kernel_launch(void* const* d_in, const int* in_sizes, int n_in,
                              void* d_out, int out_size, void* d_ws, size_t ws_size,
                              hipStream_t stream) {
  const int*   tokens = (const int*)d_in[0];
  const float* emb    = (const float*)d_in[1];
  const float* Wi_f   = (const float*)d_in[2];
  const float* Wh_f   = (const float*)d_in[3];
  const float* b_f    = (const float*)d_in[4];
  const float* Wi_b   = (const float*)d_in[5];
  const float* Wh_b   = (const float*)d_in[6];
  const float* b_b    = (const float*)d_in[7];
  const float* Wd     = (const float*)d_in[8];
  const float* bd     = (const float*)d_in[9];
  float* out = (float*)d_out;

  // workspace: emb_bf16 (8.19 MB) | wcat bf16 (1.57 MB) | feat fp32 (0.52 MB)
  char* ws = (char*)d_ws;
  unsigned short* embb = (unsigned short*)ws;
  unsigned short* wcat = (unsigned short*)(ws + 8192000);
  float*          feat = (float*)(ws + 8192000 + 1572864);

  int nemb = VOCAB * EMB;
  cvt_f32_bf16<<<(nemb + 255) / 256, 256, 0, stream>>>(emb, embb, nemb);
  int nw = NTOT * KTOT;
  pack_w<<<(nw + 255) / 256, 256, 0, stream>>>(Wi_f, Wh_f, wcat);
  pack_w<<<(nw + 255) / 256, 256, 0, stream>>>(Wi_b, Wh_b, wcat + nw);

  bilstm_kernel<<<32, 512, 0, stream>>>(tokens, embb, wcat, b_f, b_b, feat);
  head_kernel<<<(BB + 255) / 256, 256, 0, stream>>>(feat, Wd, bd, out);
}
